// MMFP4FusedMoE_63831803953558
// MI455X (gfx1250) — matrix operand order repaired and checked by reading, not compiled, not run on hardware
//
#include <hip/hip_runtime.h>

#define HDIM 2048
#define IDIM 1536
#define EDIM 16

typedef __attribute__((ext_vector_type(16))) __bf16        v16bf;
typedef __attribute__((ext_vector_type(8)))  float         v8f;
typedef __attribute__((ext_vector_type(4)))  float         v4f;
typedef __attribute__((ext_vector_type(4)))  unsigned int  v4u;
typedef __attribute__((ext_vector_type(8)))  int           v8i;
typedef __attribute__((ext_vector_type(4)))  int           v4i;

__device__ static inline unsigned short f2bf(float f) {
    unsigned int u = __float_as_uint(f);
    u += 0x7FFFu + ((u >> 16) & 1u);           // round-to-nearest-even
    return (unsigned short)(u >> 16);
}

__device__ static inline v16bf make_frag(v4u lo, v4u hi) {
    union { struct { v4u lo, hi; } p; v16bf v; } u;
    u.p.lo = lo; u.p.hi = hi;
    return u.v;
}

// ---------------------------------------------------------------------------
// TDM: async 2D bf16 tile load, global -> LDS (one descriptor per wave).
// D# layout per CDNA5 ISA 8.3/8.4: group0 = {flags, lds_addr, global_addr, type},
// group1 = {data_size, tensor dims, tile dims, stride}; remaining groups zero
// (2D tile). Tracked with TENSORcnt; completed with s_wait_tensorcnt 0.
// 6-arg builtin form (clang-23 / therock-10.0).
// ---------------------------------------------------------------------------
__device__ static inline void tdm_load_tile_bf16(
    unsigned lds_off,                 // byte offset in LDS
    const unsigned short* gptr,       // global tile start
    unsigned strideElems,             // row pitch of tensor (elements)
    unsigned tileCols,                // tile_dim0 (elements, contiguous)
    unsigned tileRows,                // tile_dim1 (rows)
    unsigned tensorD0, unsigned tensorD1) {
    unsigned long long ga = (unsigned long long)(uintptr_t)gptr;
    v4u g0;
    g0.x = 1u;                                                  // count=1 (valid), user mode
    g0.y = lds_off;                                             // lds_addr
    g0.z = (unsigned)(ga & 0xFFFFFFFFull);                      // global_addr[31:0]
    g0.w = (unsigned)((ga >> 32) & 0x1FFFFFFull) | (2u << 30);  // global_addr[56:32] | type=2
    v8i g1;
    g1[0] = (int)(1u << 16);                                    // data_size=1 -> 2 bytes
    g1[1] = (int)((tensorD0 & 0xFFFFu) << 16);                  // tensor_dim0[15:0] @ bits[63:48]
    g1[2] = (int)(((tensorD0 >> 16) & 0xFFFFu) |
                  ((tensorD1 & 0xFFFFu) << 16));                // dim0 hi | dim1 lo
    g1[3] = (int)(((tensorD1 >> 16) & 0xFFFFu) |
                  ((tileCols & 0xFFFFu) << 16));                // dim1 hi | tile_dim0
    g1[4] = (int)(tileRows & 0xFFFFu);                          // tile_dim1 | tile_dim2=0
    g1[5] = (int)strideElems;                                   // tensor_dim0_stride[31:0]
    g1[6] = 0;                                                  // stride hi16 | dim1_stride lo
    g1[7] = 0;
    v4i gz4 = {0, 0, 0, 0};
    v8i gz8 = {0, 0, 0, 0, 0, 0, 0, 0};
    __builtin_amdgcn_tensor_load_to_lds(g0, g1, gz4, gz4, gz8, 0);
}

// ---------------------------------------------------------------------------
// fp32 -> bf16 bulk convert (x is reused by 17 GEMM1 launches; convert once)
// ---------------------------------------------------------------------------
__global__ void cvt_bf16_kernel(const float* __restrict__ x,
                                unsigned short* __restrict__ xb, long n) {
    long i = ((long)blockIdx.x * blockDim.x + threadIdx.x) * 4;
    if (i + 3 < n) {
        v4f v = *(const v4f*)(x + i);
        unsigned int a = (unsigned int)f2bf(v[0]) | ((unsigned int)f2bf(v[1]) << 16);
        unsigned int b = (unsigned int)f2bf(v[2]) | ((unsigned int)f2bf(v[3]) << 16);
        *(uint2*)(xb + i) = make_uint2(a, b);
    } else {
        for (; i < n; ++i) xb[i] = f2bf(x[i]);
    }
}

// ---------------------------------------------------------------------------
// Router: logits = x @ gate_w ; top-4 ; softmax ; scatter to dense comb[T,16]
// ---------------------------------------------------------------------------
__global__ void router_kernel(const float* __restrict__ x,
                              const float* __restrict__ gw,
                              float* __restrict__ comb, int T) {
    int t = blockIdx.x * blockDim.x + threadIdx.x;
    if (t >= T) return;
    const float* xr = x + (size_t)t * HDIM;
    float lg[EDIM];
#pragma unroll
    for (int e = 0; e < EDIM; ++e) lg[e] = 0.f;
    for (int h = 0; h < HDIM; ++h) {
        float xv = xr[h];
        const float* g = gw + (size_t)h * EDIM;
#pragma unroll
        for (int e = 0; e < EDIM; ++e) lg[e] += xv * g[e];
    }
    unsigned taken = 0;
    float tv[4]; int ti[4];
    for (int k = 0; k < 4; ++k) {
        float best = -3.4e38f; int bi = 0;
#pragma unroll
        for (int e = 0; e < EDIM; ++e)
            if (!((taken >> e) & 1u) && lg[e] > best) { best = lg[e]; bi = e; }
        tv[k] = best; ti[k] = bi; taken |= 1u << bi;
    }
    float m = tv[0], s = 0.f, p[4];
    for (int k = 0; k < 4; ++k) { p[k] = __expf(tv[k] - m); s += p[k]; }
    float inv = 1.f / s;
    float row[EDIM];
#pragma unroll
    for (int e = 0; e < EDIM; ++e) row[e] = 0.f;
    for (int k = 0; k < 4; ++k) row[ti[k]] = p[k] * inv;
    float* dst = comb + (size_t)t * EDIM;
#pragma unroll
    for (int e = 0; e < EDIM; ++e) dst[e] = row[e];
}

// ---------------------------------------------------------------------------
// GEMM1 + SwiGLU:  hb = silu(xb @ Wg) * (xb @ Wu)
// Block tile 128x64, 8 waves (2M x 4N), wave tile 64x16, BK=32.
// A tiles: TDM async copy (double-buffered LDS). B tiles: reg-pipelined f32->bf16.
// ---------------------------------------------------------------------------
__global__ __launch_bounds__(256) void gemm1_act_kernel(
    const unsigned short* __restrict__ xb,   // [T, H] bf16
    const float* __restrict__ Wg,            // [H, I] f32
    const float* __restrict__ Wu,            // [H, I] f32
    unsigned short* __restrict__ hb,         // [T, I] bf16
    int T) {
    __shared__ __align__(16) unsigned short ldsX[2][128][32];
    __shared__ __align__(16) unsigned short ldsG[64][40];   // transposed [n][k], padded
    __shared__ __align__(16) unsigned short ldsU[64][40];

    const int tid  = threadIdx.x;
    const int lane = tid & 31;
    const int wave = tid >> 5;
    const int wm = wave >> 2, wn = wave & 3;
    const int lgp = lane >> 4, lc = lane & 15;

    const int col0 = blockIdx.x * 64;
    const int row0 = blockIdx.y * 128;

    const v8f zero = {0.f,0.f,0.f,0.f,0.f,0.f,0.f,0.f};
    v8f accg[4], accu[4];
#pragma unroll
    for (int i = 0; i < 4; ++i) { accg[i] = zero; accu[i] = zero; }

    // B-tile loader: thread owns k-pair {kp, kp+1} x 4 cols starting at n0
    const int kp = (tid >> 4) << 1;      // 0,2,...,30
    const int n0 = (tid & 15) * 4;       // 0..60
    v4f rg0, rg1, ru0, ru1;

    auto loadW = [&](int k0) {
        const float* pg = Wg + (size_t)(k0 + kp) * IDIM + col0 + n0;
        const float* pu = Wu + (size_t)(k0 + kp) * IDIM + col0 + n0;
        rg0 = *(const v4f*)(pg); rg1 = *(const v4f*)(pg + IDIM);
        ru0 = *(const v4f*)(pu); ru1 = *(const v4f*)(pu + IDIM);
        if (k0 + 32 < HDIM) {
            __builtin_prefetch(pg + 32 * IDIM, 0, 1);
            __builtin_prefetch(pu + 32 * IDIM, 0, 1);
        }
    };
    auto storeW = [&]() {
#pragma unroll
        for (int j = 0; j < 4; ++j) {
            unsigned pg = (unsigned)f2bf(rg0[j]) | ((unsigned)f2bf(rg1[j]) << 16);
            unsigned pu = (unsigned)f2bf(ru0[j]) | ((unsigned)f2bf(ru1[j]) << 16);
            *(unsigned*)&ldsG[n0 + j][kp] = pg;
            *(unsigned*)&ldsU[n0 + j][kp] = pu;
        }
    };
    // A-tile TDM: each wave DMAs its own 16 rows of the 128x32 tile
    auto loadA = [&](int buf, int k0) {
        unsigned lds_off = (unsigned)(uintptr_t)(void*)&ldsX[buf][wave * 16][0];
        const unsigned short* g = xb + (size_t)(row0 + wave * 16) * HDIM + k0;
        tdm_load_tile_bf16(lds_off, g, HDIM, 32, 16, HDIM, (unsigned)T);
    };

    loadA(0, 0);
    loadW(0);
    storeW();
    __builtin_amdgcn_s_wait_tensorcnt(0);
    __syncthreads();

    for (int k0 = 0; k0 < HDIM; k0 += 32) {
        const int cur = (k0 >> 5) & 1;
        const bool more = (k0 + 32) < HDIM;
        if (more) {
            loadA(cur ^ 1, k0 + 32);
            loadW(k0 + 32);
        }
        // B fragments: lane (n=lc, half=lgp) holds K = 16*lgp..16*lgp+15 contiguous
        const int bn = wn * 16 + lc;
        v16bf bg = make_frag(*(const v4u*)&ldsG[bn][16 * lgp],
                             *(const v4u*)&ldsG[bn][16 * lgp + 8]);
        v16bf bu = make_frag(*(const v4u*)&ldsU[bn][16 * lgp],
                             *(const v4u*)&ldsU[bn][16 * lgp + 8]);
#pragma unroll
        for (int i = 0; i < 4; ++i) {
            const int m = wm * 64 + i * 16 + lc;
            v16bf a = make_frag(*(const v4u*)&ldsX[cur][m][16 * lgp],
                                *(const v4u*)&ldsX[cur][m][16 * lgp + 8]);
            accg[i] = __builtin_amdgcn_wmma_f32_16x16x32_bf16(
                false, a, false, bg, (short)0, accg[i], false, false);
            accu[i] = __builtin_amdgcn_wmma_f32_16x16x32_bf16(
                false, a, false, bu, (short)0, accu[i], false, false);
        }
        __syncthreads();
        if (more) {
            storeW();
            __builtin_amdgcn_s_wait_tensorcnt(0);
            __syncthreads();
        }
    }
    // epilogue: h = silu(g)*u -> bf16.  C layout: VGPR j -> M = j + 8*half, N = lc
#pragma unroll
    for (int i = 0; i < 4; ++i) {
#pragma unroll
        for (int j = 0; j < 8; ++j) {
            float g = accg[i][j];
            float u = accu[i][j];
            float sig = __builtin_amdgcn_rcpf(1.f + __expf(-g));  // 1+e^-g >= 1: rcp safe
            float h = g * sig * u;
            int row = row0 + wm * 64 + i * 16 + 8 * lgp + j;
            int col = col0 + wn * 16 + lc;
            hb[(size_t)row * IDIM + col] = f2bf(h);
        }
    }
}

// ---------------------------------------------------------------------------
// GEMM2 + combine:  out (+)= comb[:,e] * (hb @ Wd)
// ---------------------------------------------------------------------------
__global__ __launch_bounds__(256) void gemm2_acc_kernel(
    const unsigned short* __restrict__ hb,   // [T, I] bf16
    const float* __restrict__ Wd,            // [I, H] f32
    const float* __restrict__ comb,          // [T, E] or nullptr (scale=1)
    int eIdx, int accum,
    float* __restrict__ out, int T) {
    __shared__ __align__(16) unsigned short ldsA[2][128][32];
    __shared__ __align__(16) unsigned short ldsB[64][40];

    const int tid  = threadIdx.x;
    const int lane = tid & 31;
    const int wave = tid >> 5;
    const int wm = wave >> 2, wn = wave & 3;
    const int lgp = lane >> 4, lc = lane & 15;

    const int col0 = blockIdx.x * 64;
    const int row0 = blockIdx.y * 128;

    const v8f zero = {0.f,0.f,0.f,0.f,0.f,0.f,0.f,0.f};
    v8f acc[4];
#pragma unroll
    for (int i = 0; i < 4; ++i) acc[i] = zero;

    const int kp = (tid >> 4) << 1;
    const int n0 = (tid & 15) * 4;
    v4f rw0, rw1;

    auto loadW = [&](int k0) {
        const float* pw = Wd + (size_t)(k0 + kp) * HDIM + col0 + n0;
        rw0 = *(const v4f*)(pw); rw1 = *(const v4f*)(pw + HDIM);
        if (k0 + 32 < IDIM) __builtin_prefetch(pw + 32 * HDIM, 0, 1);
    };
    auto storeW = [&]() {
#pragma unroll
        for (int j = 0; j < 4; ++j) {
            unsigned pw = (unsigned)f2bf(rw0[j]) | ((unsigned)f2bf(rw1[j]) << 16);
            *(unsigned*)&ldsB[n0 + j][kp] = pw;
        }
    };
    auto loadA = [&](int buf, int k0) {
        unsigned lds_off = (unsigned)(uintptr_t)(void*)&ldsA[buf][wave * 16][0];
        const unsigned short* g = hb + (size_t)(row0 + wave * 16) * IDIM + k0;
        tdm_load_tile_bf16(lds_off, g, IDIM, 32, 16, IDIM, (unsigned)T);
    };

    loadA(0, 0);
    loadW(0);
    storeW();
    __builtin_amdgcn_s_wait_tensorcnt(0);
    __syncthreads();

    for (int k0 = 0; k0 < IDIM; k0 += 32) {
        const int cur = (k0 >> 5) & 1;
        const bool more = (k0 + 32) < IDIM;
        if (more) {
            loadA(cur ^ 1, k0 + 32);
            loadW(k0 + 32);
        }
        const int bn = wn * 16 + lc;
        v16bf b = make_frag(*(const v4u*)&ldsB[bn][16 * lgp],
                            *(const v4u*)&ldsB[bn][16 * lgp + 8]);
#pragma unroll
        for (int i = 0; i < 4; ++i) {
            const int m = wm * 64 + i * 16 + lc;
            v16bf a = make_frag(*(const v4u*)&ldsA[cur][m][16 * lgp],
                                *(const v4u*)&ldsA[cur][m][16 * lgp + 8]);
            acc[i] = __builtin_amdgcn_wmma_f32_16x16x32_bf16(
                false, a, false, b, (short)0, acc[i], false, false);
        }
        __syncthreads();
        if (more) {
            storeW();
            __builtin_amdgcn_s_wait_tensorcnt(0);
            __syncthreads();
        }
    }
#pragma unroll
    for (int i = 0; i < 4; ++i) {
#pragma unroll
        for (int j = 0; j < 8; ++j) {
            int row = row0 + wm * 64 + i * 16 + 8 * lgp + j;
            int col = col0 + wn * 16 + lc;
            float s = comb ? comb[(size_t)row * EDIM + eIdx] : 1.0f;
            float v = s * acc[i][j];
            size_t o = (size_t)row * HDIM + col;
            out[o] = accum ? (out[o] + v) : v;
        }
    }
}

// ---------------------------------------------------------------------------
extern "C" void kernel_launch(void* const* d_in, const int* in_sizes, int n_in,
                              void* d_out, int out_size, void* d_ws, size_t ws_size,
                              hipStream_t stream) {
    const float* x  = (const float*)d_in[0];
    const float* gw = (const float*)d_in[1];
    const float* Wg = (const float*)d_in[2];
    const float* Wu = (const float*)d_in[3];
    const float* Wd = (const float*)d_in[4];
    const float* Sg = (const float*)d_in[5];
    const float* Su = (const float*)d_in[6];
    const float* Sd = (const float*)d_in[7];
    float* out = (float*)d_out;

    const int T = in_sizes[0] / HDIM;   // 4096

    // workspace layout
    char* ws = (char*)d_ws;
    unsigned short* xb = (unsigned short*)ws;                 // T*H bf16
    size_t off = ((size_t)T * HDIM * 2 + 255) & ~(size_t)255;
    unsigned short* hb = (unsigned short*)(ws + off);         // T*I bf16
    off += ((size_t)T * IDIM * 2 + 255) & ~(size_t)255;
    float* comb = (float*)(ws + off);                         // T*E f32

    long nx = (long)T * HDIM;
    cvt_bf16_kernel<<<(unsigned)((nx / 4 + 255) / 256), 256, 0, stream>>>(x, xb, nx);
    router_kernel<<<(T + 255) / 256, 256, 0, stream>>>(x, gw, comb, T);

    dim3 g1(IDIM / 64, T / 128);
    dim3 g2(HDIM / 64, T / 128);

    // shared expert first: writes out (overwrites poisoned buffer)
    gemm1_act_kernel<<<g1, 256, 0, stream>>>(xb, Sg, Su, hb, T);
    gemm2_acc_kernel<<<g2, 256, 0, stream>>>(hb, Sd, nullptr, 0, 0, out, T);

    // routed experts accumulate (sequential launches -> no races)
    for (int e = 0; e < EDIM; ++e) {
        gemm1_act_kernel<<<g1, 256, 0, stream>>>(
            xb, Wg + (size_t)e * HDIM * IDIM, Wu + (size_t)e * HDIM * IDIM, hb, T);
        gemm2_acc_kernel<<<g2, 256, 0, stream>>>(
            hb, Wd + (size_t)e * IDIM * HDIM, comb, e, 1, out, T);
    }
}
